// RN_1752346657429
// MI455X (gfx1250) — matrix operand order repaired and checked by reading, not compiled, not run on hardware
//
#include <hip/hip_runtime.h>
#include <hip/hip_bf16.h>
#include <math.h>

// ---------------------------------------------------------------------------
// Relation Network, fused for MI455X (gfx1250, wave32, WMMA).
// Dominant work (3x [262144x256]@[256x256] GEMMs, ~103 GFLOP) is fused:
// h rows are synthesized from hi/hj/hq into LDS, run through 3 bf16 WMMA
// GEMM+ReLU layers, then row-reduced with f32 atomics.
// v2: per-layer B staged into LDS via async global->LDS (ASYNCcnt).
// v3: <256 VGPRs (per-N-tile epilogue, A fragments cached per layer).
// v4: dual N-tile interleave (two independent WMMA accumulate chains ->
//     ds loads overlap WMMA execution with 2 fragments of slack) and the
//     final layer peeled out of the layer loop (no epilogue branch or
//     bias-pointer select inside the hot loop).
// ---------------------------------------------------------------------------

typedef __attribute__((ext_vector_type(16))) __bf16 v16bf;
typedef __attribute__((ext_vector_type(8)))  float  v8f;

#define MT          128   // rows of h per block
#define LDS_STRIDE  264   // bf16 elems per LDS row (256 + 8 pad, 16B aligned)

// ---------------- conv 3x3 stride2 pad1 + BN + ReLU (direct, f32) ----------
__global__ void conv_bn_relu(const float* __restrict__ in, const float* __restrict__ w,
                             const float* __restrict__ bs, const float* __restrict__ bb,
                             float* __restrict__ out,
                             int N, int Cin, int Hin, int Win, int Cout, int Hout, int Wout)
{
    int idx = blockIdx.x * blockDim.x + threadIdx.x;
    int total = N * Cout * Hout * Wout;
    if (idx >= total) return;
    int wo = idx % Wout; int t = idx / Wout;
    int ho = t % Hout;   t /= Hout;
    int co = t % Cout;   int n = t / Cout;

    float acc = 0.f;
    for (int kh = 0; kh < 3; ++kh) {
        int hin = ho * 2 + kh - 1;
        if ((unsigned)hin >= (unsigned)Hin) continue;
        for (int kw = 0; kw < 3; ++kw) {
            int win = wo * 2 + kw - 1;
            if ((unsigned)win >= (unsigned)Win) continue;
            const float* ip = in + ((size_t)(n * Cin) * Hin + hin) * Win + win;
            const float* wp = w + (size_t)((kh * 3 + kw) * Cin) * Cout + co;  // HWIO
            for (int ci = 0; ci < Cin; ++ci)
                acc += ip[(size_t)ci * Hin * Win] * wp[(size_t)ci * Cout];
        }
    }
    acc = acc * bs[co] + bb[co];
    out[idx] = fmaxf(acc, 0.f);
}

// ---------------- hi = [x|pos] @ W1a ; hj = [x|pos] @ W1b -------------------
__global__ void proj_hij(const float* __restrict__ x,     // (64,24,8,8)
                         const float* __restrict__ g_w1,  // (180,256)
                         float* __restrict__ hi, float* __restrict__ hj)
{
    int idx = blockIdx.x * blockDim.x + threadIdx.x;      // 64*64*256
    if (idx >= 64 * 64 * 256) return;
    int c = idx & 255; int t = idx >> 8;
    int p = t & 63;    int n = t >> 6;
    int h = p >> 3, wq = p & 7;
    float cx = -1.f + (2.f / 7.f) * (float)wq;
    float cy = -1.f + (2.f / 7.f) * (float)h;

    float a = 0.f, b = 0.f;
    const float* xp = x + (size_t)(n * 24) * 64 + p;
    for (int ci = 0; ci < 24; ++ci) {
        float xv = xp[(size_t)ci * 64];
        a += xv * g_w1[ci * 256 + c];
        b += xv * g_w1[(26 + ci) * 256 + c];
    }
    a += cx * g_w1[24 * 256 + c] + cy * g_w1[25 * 256 + c];
    b += cx * g_w1[(26 + 24) * 256 + c] + cy * g_w1[(26 + 25) * 256 + c];
    hi[idx] = a; hj[idx] = b;
}

// ---------------- hq = qst @ W1c + g_b1 ------------------------------------
__global__ void proj_hq(const float* __restrict__ qst, const float* __restrict__ g_w1,
                        const float* __restrict__ g_b1, float* __restrict__ hq)
{
    int idx = blockIdx.x * blockDim.x + threadIdx.x;      // 64*256
    if (idx >= 64 * 256) return;
    int c = idx & 255, n = idx >> 8;
    float a = g_b1[c];
    for (int k = 0; k < 128; ++k) a += qst[n * 128 + k] * g_w1[(52 + k) * 256 + c];
    hq[idx] = a;
}

// ------- pre-swizzle g_w2/3/4 (f32 KxN) into bf16 WMMA-B fragment order ----
// layout: [layer][nb(16)][kb(8)][lane(32)][e(16)]   (nb-major so the 8 kb
// fragments of one N-tile sit within 8 KB -> DS 16-bit immediate offsets)
//   lane: n = nb*16 + (lane&15); k = kb*32 + (lane>>4)*8 + (e&7) + (e>>3)*16
__global__ void swizzle_w(const float* __restrict__ w2, const float* __restrict__ w3,
                          const float* __restrict__ w4, __bf16* __restrict__ out)
{
    int idx = blockIdx.x * blockDim.x + threadIdx.x;      // 3*65536
    if (idx >= 3 * 65536) return;
    int layer = idx >> 16;
    int rem  = idx & 65535;
    int e    = rem & 15;
    int lane = (rem >> 4) & 31;
    int kb   = (rem >> 9) & 7;
    int nb   = (rem >> 12) & 15;
    int ncol = nb * 16 + (lane & 15);
    int k    = kb * 32 + (lane >> 4) * 8 + (e & 7) + (e >> 3) * 16;
    const float* w = (layer == 0) ? w2 : (layer == 1) ? w3 : w4;
    out[idx] = (__bf16)w[k * 256 + ncol];
}

__global__ void zero_f32(float* __restrict__ p, int n)
{
    int i = blockIdx.x * blockDim.x + threadIdx.x;
    if (i < n) p[i] = 0.f;
}

// ---------------- fused relational core: 3 GEMM+ReLU layers + row reduce ---
__global__ __launch_bounds__(256, 1)
void rn_gemm_fused(const float* __restrict__ hi, const float* __restrict__ hj,
                   const float* __restrict__ hq, const __bf16* __restrict__ wswz,
                   const float* __restrict__ b2, const float* __restrict__ b3,
                   const float* __restrict__ b4, float* __restrict__ sbuf)
{
    __shared__ __align__(16) __bf16 lds_h[MT * LDS_STRIDE];   // 67584 B
    __shared__ __align__(32) __bf16 lds_b[65536];             // 131072 B (one layer of B)

    const int tid   = threadIdx.x;
    const int block = blockIdx.x;           // 0..2047
    const int n     = block >> 5;           // 32 blocks per batch image
    const int t0    = (block & 31) * MT;    // first (i,j) row of this tile

    const unsigned lds_b_base = (unsigned)(size_t)&lds_b[0];

    // async stage of one layer's swizzled B (128 KB) into LDS.
    // Each lane copies 16 B per op: 256 thr * 32 iters * 16 B = 131072 B.
    auto async_copy_B = [&](int layer) {
        const char* src = (const char*)(wswz + (size_t)layer * 65536);
        #pragma unroll 4
        for (int i = 0; i < 32; ++i) {
            unsigned loff = lds_b_base + (unsigned)(tid * 16 + i * 4096);
            unsigned long long ga = (unsigned long long)(src + tid * 16 + i * 4096);
            asm volatile("global_load_async_to_lds_b128 %0, %1, off"
                         :: "v"(loff), "v"(ga) : "memory");
        }
    };

    // Kick off layer-0 B copy so it overlaps the A-tile build
    async_copy_B(0);

    // Build h0 tile: relu(hi[n,i] + hj[n,j] + hq[n]) -> bf16 in LDS
    for (int e = tid; e < MT * 256; e += 256) {
        int r = e >> 8, c = e & 255;
        int rr = t0 + r;                    // 0..4095 within batch n
        int i = rr >> 6, j = rr & 63;
        float v = hi[(size_t)(n * 64 + i) * 256 + c]
                + hj[(size_t)(n * 64 + j) * 256 + c]
                + hq[n * 256 + c];
        lds_h[r * LDS_STRIDE + c] = (__bf16)fmaxf(v, 0.f);
    }

    const int wave  = tid >> 5;             // 0..7 -> 16-row M slab (wave-private)
    const int lane  = tid & 31;
    const int mbase = wave * 16;
    const int lrow  = mbase + (lane & 15);  // A-matrix row this lane reads
    const int sel   = (lane < 16) ? 0 : 8;  // K sub-select per 16-bit A layout
    const int mhalf = (lane < 16) ? 0 : 8;  // C/D layout: lanes 16-31 hold M=8..15

    // Cache this wave's 16 rows of A (whole K=256): 8 frags = 64 VGPRs,
    // reused across all 16 N-tiles of a layer.
    v16bf afrag[8];
    auto load_A = [&]() {
        #pragma unroll
        for (int kb = 0; kb < 8; ++kb) {
            union { v16bf v; uint4 q[2]; } ua;
            const __bf16* rp = &lds_h[lrow * LDS_STRIDE + kb * 32];
            ua.q[0] = *(const uint4*)(rp + sel);
            ua.q[1] = *(const uint4*)(rp + 16 + sel);
            afrag[kb] = ua.v;
        }
    };

    // Dual-N-tile GEMM core: two independent accumulate chains share afrag;
    // per k-step 4 ds_load_b128 feed 2 WMMAs, so loads overlap WMMA exec.
    auto gemm_pair = [&](int nb0, v8f& a0, v8f& a1) {
        const __bf16* bb0 = &lds_b[(size_t)nb0 * 4096];
        const __bf16* bb1 = bb0 + 4096;
        a0 = (v8f){0.f, 0.f, 0.f, 0.f, 0.f, 0.f, 0.f, 0.f};
        a1 = (v8f){0.f, 0.f, 0.f, 0.f, 0.f, 0.f, 0.f, 0.f};
        #pragma unroll
        for (int kb = 0; kb < 8; ++kb) {
            v16bf f0 = *(const v16bf*)&bb0[(kb * 32 + lane) * 16];
            v16bf f1 = *(const v16bf*)&bb1[(kb * 32 + lane) * 16];
            a0 = __builtin_amdgcn_wmma_f32_16x16x32_bf16(
                false, afrag[kb], false, f0, (short)0, a0, false, false);
            a1 = __builtin_amdgcn_wmma_f32_16x16x32_bf16(
                false, afrag[kb], false, f1, (short)0, a1, false, false);
        }
    };

    // ---------------- layers 0,1: GEMM -> bias+relu -> bf16 back to LDS ----
    for (int layer = 0; layer < 2; ++layer) {
        asm volatile("s_wait_asynccnt 0x0" ::: "memory");
        __syncthreads();
        load_A();
        const float* bias = layer ? b3 : b2;

        #pragma unroll 1
        for (int nbp = 0; nbp < 8; ++nbp) {
            int nb0 = nbp * 2;
            v8f a0, a1;
            gemm_pair(nb0, a0, a1);

            int c0 = nb0 * 16 + (lane & 15);
            float bv0 = bias[c0], bv1 = bias[c0 + 16];
            #pragma unroll
            for (int r = 0; r < 8; ++r) {
                int M = mbase + mhalf + r;
                lds_h[M * LDS_STRIDE + c0]      = (__bf16)fmaxf(a0[r] + bv0, 0.f);
                lds_h[M * LDS_STRIDE + c0 + 16] = (__bf16)fmaxf(a1[r] + bv1, 0.f);
            }
        }
        // all waves must be done reading lds_b before we overwrite it
        __syncthreads();
        async_copy_B(layer + 1);
    }

    // ---------------- final layer: GEMM -> bias+relu -> row reduce ---------
    asm volatile("s_wait_asynccnt 0x0" ::: "memory");
    __syncthreads();
    load_A();

    #pragma unroll 1
    for (int nbp = 0; nbp < 8; ++nbp) {
        int nb0 = nbp * 2;
        v8f a0, a1;
        gemm_pair(nb0, a0, a1);

        int c0 = nb0 * 16 + (lane & 15);
        float bv0 = b4[c0], bv1 = b4[c0 + 16];
        float p0 = 0.f, p1 = 0.f;
        #pragma unroll
        for (int r = 0; r < 8; ++r) {
            p0 += fmaxf(a0[r] + bv0, 0.f);
            p1 += fmaxf(a1[r] + bv1, 0.f);
        }
        // lanes L and L+16 hold the same N column -> combine halves
        p0 += __shfl_xor(p0, 16, 32);
        p1 += __shfl_xor(p1, 16, 32);
        if (lane < 16) {
            atomicAdd(&sbuf[n * 256 + c0], p0);
            atomicAdd(&sbuf[n * 256 + c0 + 16], p1);
        }
    }
}

// ---------------- head: 2 FC+ReLU, classifier, log_softmax -----------------
__global__ __launch_bounds__(256)
void head(const float* __restrict__ sbuf,
          const float* __restrict__ f_w1, const float* __restrict__ f_b1,
          const float* __restrict__ f_w2, const float* __restrict__ f_b2,
          const float* __restrict__ cls_w, const float* __restrict__ cls_b,
          float* __restrict__ out)
{
    __shared__ float sh[256], y1[256], y2[256], lg[32], red[2];
    int n = blockIdx.x, t = threadIdx.x;
    sh[t] = sbuf[n * 256 + t];
    __syncthreads();
    float a = f_b1[t];
    for (int k = 0; k < 256; ++k) a += sh[k] * f_w1[k * 256 + t];
    y1[t] = fmaxf(a, 0.f);
    __syncthreads();
    a = f_b2[t];
    for (int k = 0; k < 256; ++k) a += y1[k] * f_w2[k * 256 + t];
    y2[t] = fmaxf(a, 0.f);
    __syncthreads();
    if (t < 32) {
        a = cls_b[t];
        for (int k = 0; k < 256; ++k) a += y2[k] * cls_w[k * 32 + t];
        lg[t] = a;
    }
    __syncthreads();
    if (t == 0) {
        float m = lg[0];
        for (int k = 1; k < 32; ++k) m = fmaxf(m, lg[k]);
        float s = 0.f;
        for (int k = 0; k < 32; ++k) s += __expf(lg[k] - m);
        red[0] = m; red[1] = logf(s);
    }
    __syncthreads();
    if (t < 32) out[n * 32 + t] = lg[t] - red[0] - red[1];
}

// ---------------------------------------------------------------------------
extern "C" void kernel_launch(void* const* d_in, const int* in_sizes, int n_in,
                              void* d_out, int out_size, void* d_ws, size_t ws_size,
                              hipStream_t stream)
{
    const float* img    = (const float*)d_in[0];
    const float* qst    = (const float*)d_in[1];
    const float* cw[4]  = {(const float*)d_in[2], (const float*)d_in[3],
                           (const float*)d_in[4], (const float*)d_in[5]};
    const float* bns[4] = {(const float*)d_in[6], (const float*)d_in[8],
                           (const float*)d_in[10], (const float*)d_in[12]};
    const float* bnb[4] = {(const float*)d_in[7], (const float*)d_in[9],
                           (const float*)d_in[11], (const float*)d_in[13]};
    const float* g_w1 = (const float*)d_in[14];
    const float* g_b1 = (const float*)d_in[15];
    const float* g_w2 = (const float*)d_in[16];
    const float* g_b2 = (const float*)d_in[17];
    const float* g_w3 = (const float*)d_in[18];
    const float* g_b3 = (const float*)d_in[19];
    const float* g_w4 = (const float*)d_in[20];
    const float* g_b4 = (const float*)d_in[21];
    const float* f_w1 = (const float*)d_in[22];
    const float* f_b1 = (const float*)d_in[23];
    const float* f_w2 = (const float*)d_in[24];
    const float* f_b2 = (const float*)d_in[25];
    const float* clsw = (const float*)d_in[26];
    const float* clsb = (const float*)d_in[27];
    float* out = (float*)d_out;

    // workspace layout (bytes)
    char* ws = (char*)d_ws;
    float*  c0   = (float*)(ws + 0);          // 64*24*64*64  = 25165824 B
    float*  c1   = (float*)(ws + 25165824);   // 64*24*32*32  =  6291456 B
    float*  c2   = (float*)(ws + 31457280);   // 64*24*16*16  =  1572864 B
    float*  c3   = (float*)(ws + 33030144);   // 64*24*8*8    =   393216 B
    float*  hi   = (float*)(ws + 33423360);   // 64*64*256    =  4194304 B
    float*  hj   = (float*)(ws + 37617664);   // 64*64*256    =  4194304 B
    float*  hq   = (float*)(ws + 41811968);   // 64*256       =    65536 B
    float*  sbuf = (float*)(ws + 41877504);   // 64*256       =    65536 B
    __bf16* wswz = (__bf16*)(ws + 41943040);  // 3*65536 bf16 =   393216 B
    // total ~42.3 MB

    zero_f32<<<64, 256, 0, stream>>>(sbuf, 64 * 256);
    swizzle_w<<<768, 256, 0, stream>>>(g_w2, g_w3, g_w4, wswz);

    conv_bn_relu<<<24576, 256, 0, stream>>>(img, cw[0], bns[0], bnb[0], c0,
                                            64, 3, 128, 128, 24, 64, 64);
    conv_bn_relu<<<6144, 256, 0, stream>>>(c0, cw[1], bns[1], bnb[1], c1,
                                           64, 24, 64, 64, 24, 32, 32);
    conv_bn_relu<<<1536, 256, 0, stream>>>(c1, cw[2], bns[2], bnb[2], c2,
                                           64, 24, 32, 32, 24, 16, 16);
    conv_bn_relu<<<384, 256, 0, stream>>>(c2, cw[3], bns[3], bnb[3], c3,
                                          64, 24, 16, 16, 24, 8, 8);

    proj_hij<<<4096, 256, 0, stream>>>(c3, g_w1, hi, hj);
    proj_hq<<<64, 256, 0, stream>>>(qst, g_w1, g_b1, hq);

    rn_gemm_fused<<<2048, 256, 0, stream>>>(hi, hj, hq, wswz, g_b2, g_b3, g_b4, sbuf);

    head<<<64, 256, 0, stream>>>(sbuf, f_w1, f_b1, f_w2, f_b2, clsw, clsb, out);
}
